// VQ_VAE_z32_8924942041590
// MI455X (gfx1250) — matrix operand order repaired and checked by reading, not compiled, not run on hardware
//
#include <hip/hip_runtime.h>

typedef __attribute__((ext_vector_type(16))) _Float16 v16h;
typedef __attribute__((ext_vector_type(8)))  _Float16 v8h;
typedef __attribute__((ext_vector_type(8)))  float    v8f;
typedef __attribute__((ext_vector_type(4)))  int      v4i;

// --------------------------------------------------------------------
// CDNA5 async global->LDS copy (ASYNCcnt path) with safe fallback.
// Probe round 2 showed the builtin takes int4* (AS-qualified) pointers.
// --------------------------------------------------------------------
#define USE_ASYNC_LDS_DEF 0
#if defined(__has_builtin)
#if __has_builtin(__builtin_amdgcn_global_load_async_to_lds_b128) && \
    __has_builtin(__builtin_amdgcn_s_wait_asynccnt)
#undef USE_ASYNC_LDS_DEF
#define USE_ASYNC_LDS_DEF 1
#endif
#endif

#if USE_ASYNC_LDS_DEF
typedef __attribute__((address_space(1))) v4i* as1_v4i_p;  // global int4*
typedef __attribute__((address_space(3))) v4i* as3_v4i_p;  // LDS int4*
#define ASYNC_CP16(dst, src)                                              \
  __builtin_amdgcn_global_load_async_to_lds_b128(                         \
      (as1_v4i_p)(src), (as3_v4i_p)(dst), 0, 0)
#define ASYNC_WAIT() __builtin_amdgcn_s_wait_asynccnt(0)
#else
#define ASYNC_CP16(dst, src) (*(v8h*)(dst) = *(const v8h*)(src))
#define ASYNC_WAIT() ((void)0)
#endif

// ---------------- model dims ----------------
static constexpr int  BN_  = 64;     // batch
static constexpr int  CIN_ = 2;
static constexpr int  IMG_ = 128;
static constexpr int  HC_  = 64;     // hidden channels
static constexpr int  RC_  = 128;    // residual hidden channels
static constexpr int  KC_  = 512;    // codebook entries
static constexpr long M1_  = (long)BN_ * 64 * 64;   // 262144 rows after stride-2 #1
static constexpr long M2_  = (long)BN_ * 32 * 32;   // 65536 rows after stride-2 #2
static constexpr float EPS_ = 1e-5f;

// ---------------- workspace layout (bytes) ----------------
static constexpr size_t OFF_ACT0  = 0;           // [M1,32] f32 NHWC
static constexpr size_t OFF_ACT1  = 33554432;    // [M2,64] f32 (z / enc act)
static constexpr size_t OFF_ACT2  = 50331648;    // [M2,128] f32 (res mid)
static constexpr size_t OFF_ACT3  = 83886080;    // [M2,64] f32 (res h)
static constexpr size_t OFF_ACT4  = 100663296;   // [M2,64] f32 (quant / dec act)
static constexpr size_t OFF_ACT5  = 117440512;   // [M1,32] f32 (deconv1 out)
static constexpr size_t OFF_COL   = 150994944;   // f16 im2col panel (max 65536x576), also ZH
static constexpr size_t OFF_WH    = 226492416;   // f16 weight panel [N,K]
static constexpr size_t OFF_CBH   = 227540992;   // f16 codebook [512,64]
static constexpr size_t OFF_IDX   = 227606528;   // int [65536]
static constexpr size_t OFF_STATS = 227868672;   // f32 [256] sum/sumsq
static constexpr size_t OFF_CNT   = 227869696;   // f32 [512] counts
static constexpr size_t OFF_CBN   = 227871744;   // f32 [512] ||e||^2
static constexpr size_t OFF_ACC   = 227873792;   // f32 [2]: vq-mse, recon

// =====================================================================
// elementwise helpers
// =====================================================================
__global__ void zero_f32_k(float* __restrict__ p, long n) {
  long i = (long)blockIdx.x * blockDim.x + threadIdx.x;
  long st = (long)gridDim.x * blockDim.x;
  for (; i < n; i += st) p[i] = 0.f;
}

__global__ void cvt_f16_k(const float* __restrict__ x, _Float16* __restrict__ y,
                          long n, int relu) {
  long i = (long)blockIdx.x * blockDim.x + threadIdx.x;
  long st = (long)gridDim.x * blockDim.x;
  for (; i < n; i += st) {
    float v = x[i];
    if (relu) v = fmaxf(v, 0.f);
    y[i] = (_Float16)v;
  }
}

// im2col: x (NCHW or NHWC f32) -> col [M = B*OH*OW, K = Cin*kh*kw] f16
__global__ void im2col_k(const float* __restrict__ x, _Float16* __restrict__ col,
                         int Bn, int Cin, int IH, int IW, int OH, int OW,
                         int kh, int kw, int stride, int pad, int src_nchw, int relu) {
  const int Kd = Cin * kh * kw;
  const long total = (long)Bn * OH * OW * Kd;
  long i = (long)blockIdx.x * blockDim.x + threadIdx.x;
  long st = (long)gridDim.x * blockDim.x;
  for (; i < total; i += st) {
    int kk = (int)(i % Kd);
    long m = i / Kd;
    int ci = kk / (kh * kw);
    int r  = kk - ci * kh * kw;
    int ky = r / kw, kx = r - ky * kw;
    int ox = (int)(m % OW);
    long t = m / OW;
    int oy = (int)(t % OH);
    int b  = (int)(t / OH);
    int iy = oy * stride - pad + ky;
    int ix = ox * stride - pad + kx;
    float v = 0.f;
    if (iy >= 0 && iy < IH && ix >= 0 && ix < IW) {
      v = src_nchw ? x[(((long)b * Cin + ci) * IH + iy) * IW + ix]
                   : x[(((long)b * IH + iy) * IW + ix) * Cin + ci];
      if (relu) v = fmaxf(v, 0.f);
    }
    col[i] = (_Float16)v;
  }
}

// =====================================================================
// WMMA GEMM tile helpers
// =====================================================================
union AVu { v16h v; v8h h[2]; };

__device__ __forceinline__ void load_a4(AVu* a, const _Float16* __restrict__ ap0,
                                        size_t rstride, int k) {
#pragma unroll
  for (int t = 0; t < 4; ++t) {
    const _Float16* ap = ap0 + t * rstride;
    a[t].h[0] = *(const v8h*)(ap + k);       // K {kt..kt+7} / {kt+8..kt+15}
    a[t].h[1] = *(const v8h*)(ap + k + 16);  // K {kt+16..}  / {kt+24..}
  }
}

__device__ __forceinline__ void wmma4(v8f* acc, const AVu* a,
                                      const _Float16* bl, int k) {
  AVu b;                                     // bl points into LDS -> ds_load
  b.h[0] = *(const v8h*)(bl + k);
  b.h[1] = *(const v8h*)(bl + k + 16);
#pragma unroll
  for (int t = 0; t < 4; ++t)
    acc[t] = __builtin_amdgcn_wmma_f32_16x16x32_f16(
        false, a[t].v, false, b.v, (short)0, acc[t], false, false);
}

// =====================================================================
// WMMA GEMM: C[m,n] = sum_k A[m,k]*B[n,k] + bias[n]
// A: [M,K] f16 row-major (im2col), B: [N,K] f16 row-major (OIHW flat)
// Block = 4 waves; each wave owns a 64x16 tile; all waves share one
// Nbase so the B panel is staged once per block in LDS (async DMA when
// available). A stream is double-buffered in registers + prefetched.
// Requires: M % 256 == 0, K % 32 == 0, K <= 576.
// =====================================================================
__global__ __launch_bounds__(128)
void gemm_wmma_bias(const _Float16* __restrict__ A, const _Float16* __restrict__ Bw,
                    const float* __restrict__ bias, float* __restrict__ C,
                    long M, int N, int K) {
  __shared__ _Float16 bpan[16 * 584];        // [16][K+8] padded vs bank conflicts
  const int tid  = threadIdx.x;
  const int lane = tid & 31;
  const int wave = tid >> 5;
  const int hi   = lane >> 4;                // lane half selects K sub-group
  const int l16  = lane & 15;
  const int ldb  = K + 8;
  const int Nbase = blockIdx.y * 16;

  // cooperative B panel fill: global -> LDS
  {
    const int cpr = K >> 3;                  // 16B chunks per row
    const int total = 16 * cpr;
    for (int c = tid; c < total; c += 128) {
      int row = c / cpr;
      int ch  = c - row * cpr;
      const _Float16* src = Bw + (size_t)(Nbase + row) * K + ch * 8;
      _Float16* dst = bpan + row * ldb + ch * 8;
      ASYNC_CP16(dst, src);
    }
    ASYNC_WAIT();
    __syncthreads();
  }

  const long Mbase = ((long)blockIdx.x * 4 + wave) * 64;
  const _Float16* ap0 = A + (size_t)(Mbase + l16) * K + hi * 8;
  const size_t rstride = (size_t)K * 16;
  const _Float16* bl = bpan + l16 * ldb + hi * 8;

  v8f acc[4] = {{}, {}, {}, {}};
  AVu aA[4], aB[4];
  load_a4(aA, ap0, rstride, 0);
  int k = 0;
  for (; k + 64 < K; k += 64) {              // ping-pong double buffer
    load_a4(aB, ap0, rstride, k + 32);
    __builtin_prefetch((const void*)(ap0 + k + 128), 0, 3);
    wmma4(acc, aA, bl, k);
    load_a4(aA, ap0, rstride, k + 64);
    wmma4(acc, aB, bl, k + 32);
  }
  if (k + 32 < K) {                          // 2-tile tail
    load_a4(aB, ap0, rstride, k + 32);
    wmma4(acc, aA, bl, k);
    wmma4(acc, aB, bl, k + 32);
  } else {                                   // 1-tile tail (K == 32)
    wmma4(acc, aA, bl, k);
  }

  const float bv = bias ? bias[Nbase + l16] : 0.f;
#pragma unroll
  for (int t = 0; t < 4; ++t) {
#pragma unroll
    for (int r = 0; r < 8; ++r) {
      long mrow = Mbase + t * 16 + (hi ? r + 8 : r);  // D layout: lane half = M+8
      C[(size_t)mrow * N + Nbase + l16] = acc[t][r] + bv;
    }
  }
}

// =====================================================================
// Fused VQ nearest-codebook search. Block = 4 waves = 64 z rows.
// Whole f16 codebook (512x64) staged in LDS (72-half padded rows),
// then each wave runs 2 WMMAs per 16-code tile from LDS, tracks
// per-lane min/argmin, and xor-reduces within 16-lane groups.
// =====================================================================
__global__ __launch_bounds__(128)
void vq_argmin_k(const _Float16* __restrict__ ZH,   // [M,64] f16
                 const _Float16* __restrict__ CBH,  // [512,64] f16
                 const float* __restrict__ cbn,     // [512]
                 int* __restrict__ idxOut) {
  __shared__ _Float16 cpan[512 * 72];        // 72 KB codebook panel
  const int tid = threadIdx.x;
  {
    for (int c = tid; c < 512 * 8; c += 128) {
      int row = c >> 3;
      int ch  = c & 7;
      const _Float16* src = CBH + (size_t)row * 64 + ch * 8;
      _Float16* dst = cpan + row * 72 + ch * 8;
      ASYNC_CP16(dst, src);
    }
    ASYNC_WAIT();
    __syncthreads();
  }

  const int lane = tid & 31;
  const int wave = tid >> 5;
  const int hi   = lane >> 4;
  const int l16  = lane & 15;
  const long Mbase = (long)blockIdx.x * 64 + wave * 16;

  const _Float16* ap = ZH + (size_t)(Mbase + l16) * 64 + hi * 8;
  AVu a0, a1;
  a0.h[0] = *(const v8h*)(ap);
  a0.h[1] = *(const v8h*)(ap + 16);
  a1.h[0] = *(const v8h*)(ap + 32);
  a1.h[1] = *(const v8h*)(ap + 48);

  float best[8];
  int   bidx[8];
#pragma unroll
  for (int r = 0; r < 8; ++r) { best[r] = 1e30f; bidx[r] = 0; }

  for (int n = 0; n < KC_; n += 16) {
    const _Float16* bp = cpan + (size_t)(n + l16) * 72 + hi * 8;  // LDS
    AVu b0, b1;
    b0.h[0] = *(const v8h*)(bp);
    b0.h[1] = *(const v8h*)(bp + 16);
    b1.h[0] = *(const v8h*)(bp + 32);
    b1.h[1] = *(const v8h*)(bp + 48);
    v8f acc = {};
    acc = __builtin_amdgcn_wmma_f32_16x16x32_f16(false, a0.v, false, b0.v, (short)0, acc, false, false);
    acc = __builtin_amdgcn_wmma_f32_16x16x32_f16(false, a1.v, false, b1.v, (short)0, acc, false, false);
    const float cn = cbn[n + l16];
    const int col = n + l16;
#pragma unroll
    for (int r = 0; r < 8; ++r) {
      float sc = cn - 2.f * acc[r];
      if (sc < best[r]) { best[r] = sc; bidx[r] = col; }
    }
  }
#pragma unroll
  for (int off = 8; off >= 1; off >>= 1) {
#pragma unroll
    for (int r = 0; r < 8; ++r) {
      float ob = __shfl_xor(best[r], off, 16);
      int   oi = __shfl_xor(bidx[r], off, 16);
      if (ob < best[r] || (ob == best[r] && oi < bidx[r])) { best[r] = ob; bidx[r] = oi; }
    }
  }
  if (l16 == 0) {
#pragma unroll
    for (int r = 0; r < 8; ++r)
      idxOut[Mbase + (hi ? r + 8 : r)] = bidx[r];
  }
}

// =====================================================================
// BatchNorm (training-mode batch stats) on NHWC-flattened [M,C]
// Fixed channel per thread: coalesced reads, register accumulation,
// only 2 LDS atomics per thread. Requires blockDim % C == 0.
// =====================================================================
__global__ __launch_bounds__(256)
void bn_stats_k(const float* __restrict__ x, float* __restrict__ sums, long M, int C) {
  __shared__ float ls[256];                  // [C] sum, [C] sumsq, C<=128
  const int tid = threadIdx.x;
  for (int i = tid; i < 2 * C; i += blockDim.x) ls[i] = 0.f;
  __syncthreads();
  const int rowsPerBlk = blockDim.x / C;
  const int c = tid % C;
  long m = (long)blockIdx.x * rowsPerBlk + tid / C;
  const long mstep = (long)gridDim.x * rowsPerBlk;
  float s = 0.f, s2 = 0.f;
  for (; m < M; m += mstep) {
    float v = x[m * C + c];
    s += v;
    s2 += v * v;
  }
  atomicAdd(&ls[c], s);
  atomicAdd(&ls[C + c], s2);
  __syncthreads();
  for (int i = tid; i < 2 * C; i += blockDim.x) atomicAdd(&sums[i], ls[i]);
}

__global__ void bn_apply_k(const float* __restrict__ x, float* __restrict__ y,
                           const float* __restrict__ res, const float* __restrict__ sums,
                           const float* __restrict__ g, const float* __restrict__ bb,
                           long M, int C, int relu) {
  const long total = M * C;
  long i = (long)blockIdx.x * blockDim.x + threadIdx.x;
  long st = (long)gridDim.x * blockDim.x;
  const float invM = 1.f / (float)M;
  for (; i < total; i += st) {
    int c = (int)(i % C);
    float mean = sums[c] * invM;
    float var  = sums[C + c] * invM - mean * mean;
    float v = g[c] * (x[i] - mean) * rsqrtf(var + EPS_) + bb[c];
    if (res) v += res[i];
    if (relu) v = fmaxf(v, 0.f);
    y[i] = v;
  }
}

// =====================================================================
// Direct ConvTranspose2d(k=4,s=2,p=1): in NHWC, w [Ci,Co,4,4], out NHWC/NCHW
// =====================================================================
__global__ void deconv_k(const float* __restrict__ x, const float* __restrict__ w,
                         const float* __restrict__ bias, float* __restrict__ y,
                         int Bn, int IH, int IW, int Ci, int Co, int out_nchw) {
  const int OH = IH * 2, OW = IW * 2;
  const long total = (long)Bn * OH * OW * Co;
  long i = (long)blockIdx.x * blockDim.x + threadIdx.x;
  long st = (long)gridDim.x * blockDim.x;
  for (; i < total; i += st) {
    int co, oy, ox, b;
    if (out_nchw) {
      ox = (int)(i % OW); long t = i / OW;
      oy = (int)(t % OH); t /= OH;
      co = (int)(t % Co); b = (int)(t / Co);
    } else {
      co = (int)(i % Co); long t = i / Co;
      ox = (int)(t % OW); t /= OW;
      oy = (int)(t % OH); b = (int)(t / OH);
    }
    float s = bias[co];
    for (int ky = 0; ky < 4; ++ky) {
      int iy2 = oy + 1 - ky;
      if (iy2 < 0 || (iy2 & 1)) continue;
      int iy = iy2 >> 1;
      if (iy >= IH) continue;
      for (int kx = 0; kx < 4; ++kx) {
        int ix2 = ox + 1 - kx;
        if (ix2 < 0 || (ix2 & 1)) continue;
        int ix = ix2 >> 1;
        if (ix >= IW) continue;
        const float* xp = x + (((long)b * IH + iy) * IW + ix) * Ci;
        const float* wp = w + ((size_t)co * 4 + ky) * 4 + kx;
        for (int ci = 0; ci < Ci; ++ci)
          s += xp[ci] * wp[(size_t)ci * Co * 16];
      }
    }
    y[i] = s;  // i already enumerates the chosen output layout linearly
  }
}

// =====================================================================
// VQ support + losses
// =====================================================================
__global__ void cb_norm_k(const float* __restrict__ cb, float* __restrict__ cbn,
                          int Kc, int C) {
  int n = blockIdx.x * blockDim.x + threadIdx.x;
  if (n < Kc) {
    float s = 0.f;
    for (int c = 0; c < C; ++c) { float v = cb[(size_t)n * C + c]; s += v * v; }
    cbn[n] = s;
  }
}

__global__ void vq_gather_k(const float* __restrict__ z, const float* __restrict__ cb,
                            const int* __restrict__ idx, float* __restrict__ q,
                            float* __restrict__ counts, float* __restrict__ acc,
                            long M, int C) {
  __shared__ float ls;
  if (threadIdx.x == 0) ls = 0.f;
  __syncthreads();
  const long total = M * C;
  long i = (long)blockIdx.x * blockDim.x + threadIdx.x;
  long st = (long)gridDim.x * blockDim.x;
  float p = 0.f;
  for (; i < total; i += st) {
    long m = i / C;
    int c = (int)(i % C);
    int id = idx[m];
    float qv = cb[(size_t)id * C + c];
    float d = qv - z[i];
    p += d * d;
    q[i] = qv;
    if (c == 0) atomicAdd(&counts[id], 1.f);
  }
  atomicAdd(&ls, p);
  __syncthreads();
  if (threadIdx.x == 0) atomicAdd(acc, ls);
}

__global__ void recon_loss_k(const float* __restrict__ dec, const float* __restrict__ inp,
                             const float* __restrict__ cvar, float* __restrict__ acc,
                             long n, int C, long HW) {
  __shared__ float ls;
  if (threadIdx.x == 0) ls = 0.f;
  __syncthreads();
  long i = (long)blockIdx.x * blockDim.x + threadIdx.x;
  long st = (long)gridDim.x * blockDim.x;
  float p = 0.f;
  for (; i < n; i += st) {
    int c = (int)((i / HW) % C);
    float d = dec[i] - inp[i];
    p += d * d / cvar[c];
  }
  atomicAdd(&ls, p);
  __syncthreads();
  if (threadIdx.x == 0) atomicAdd(acc, ls);
}

__global__ __launch_bounds__(512)
void finalize_k(const float* __restrict__ counts, const float* __restrict__ acc,
                float* __restrict__ tail) {
  __shared__ float sh[512];
  int t = threadIdx.x;
  float p = counts[t] / (float)M2_;
  sh[t] = p * logf(p + 1e-10f);
  __syncthreads();
  for (int s = 256; s > 0; s >>= 1) {
    if (t < s) sh[t] += sh[t + s];
    __syncthreads();
  }
  if (t == 0) {
    float perp   = expf(-sh[0]);
    float mse    = acc[0] / (float)(M2_ * HC_);
    float closs  = 1.25f * mse;  // q_loss + BETA*e_loss, both == mse in fwd
    float recon  = acc[1] / (float)((long)BN_ * CIN_ * IMG_ * IMG_);
    tail[0] = recon;
    tail[1] = closs;
    tail[2] = perp;
    tail[3] = recon + closs;
  }
}

// =====================================================================
// host side
// =====================================================================
static inline unsigned nblk(long n, int tpb, long cap = 32768) {
  long b = (n + tpb - 1) / tpb;
  if (b > cap) b = cap;
  if (b < 1) b = 1;
  return (unsigned)b;
}

static void run_resblock(hipStream_t s, float* act, float* tmpR, float* tmpH,
                         _Float16* col, _Float16* wh, float* stats,
                         const float* w1, const float* b1, const float* g1, const float* bb1,
                         const float* w2, const float* b2, const float* g2, const float* bb2) {
  const long M = M2_;
  const int K1 = HC_ * 9;  // 576
  // conv3x3 HC->RC on relu(act)
  cvt_f16_k<<<nblk((long)RC_ * K1, 256), 256, 0, s>>>(w1, wh, (long)RC_ * K1, 0);
  im2col_k<<<nblk(M * K1, 256), 256, 0, s>>>(act, col, BN_, HC_, 32, 32, 32, 32,
                                             3, 3, 1, 1, 0, 1);
  gemm_wmma_bias<<<dim3((unsigned)(M / 256), RC_ / 16), 128, 0, s>>>(col, wh, b1, tmpR, M, RC_, K1);
  zero_f32_k<<<1, 256, 0, s>>>(stats, 2 * RC_);
  bn_stats_k<<<2048, 256, 0, s>>>(tmpR, stats, M, RC_);
  bn_apply_k<<<nblk(M * RC_, 256), 256, 0, s>>>(tmpR, tmpR, nullptr, stats, g1, bb1, M, RC_, 1);
  // conv1x1 RC->HC (input already relu'd above)
  cvt_f16_k<<<nblk(M * RC_, 256), 256, 0, s>>>(tmpR, col, M * RC_, 0);
  cvt_f16_k<<<nblk((long)HC_ * RC_, 256), 256, 0, s>>>(w2, wh, (long)HC_ * RC_, 0);
  gemm_wmma_bias<<<dim3((unsigned)(M / 256), HC_ / 16), 128, 0, s>>>(col, wh, b2, tmpH, M, HC_, RC_);
  zero_f32_k<<<1, 256, 0, s>>>(stats, 2 * HC_);
  bn_stats_k<<<2048, 256, 0, s>>>(tmpH, stats, M, HC_);
  bn_apply_k<<<nblk(M * HC_, 256), 256, 0, s>>>(tmpH, act, act, stats, g2, bb2, M, HC_, 0);
}

extern "C" void kernel_launch(void* const* d_in, const int* in_sizes, int n_in,
                              void* d_out, int out_size, void* d_ws, size_t ws_size,
                              hipStream_t stream) {
  const float* inputs   = (const float*)d_in[0];
  const float* enc_w1   = (const float*)d_in[1];
  const float* enc_b1   = (const float*)d_in[2];
  const float* enc_g1   = (const float*)d_in[3];
  const float* enc_bb1  = (const float*)d_in[4];
  const float* enc_w2   = (const float*)d_in[5];
  const float* enc_b2   = (const float*)d_in[6];
  const float* enc_g2   = (const float*)d_in[7];
  const float* enc_bb2  = (const float*)d_in[8];
  const float* erw1     = (const float*)d_in[9];
  const float* erb1     = (const float*)d_in[10];
  const float* erg1     = (const float*)d_in[11];
  const float* erbb1    = (const float*)d_in[12];
  const float* erw2     = (const float*)d_in[13];
  const float* erb2     = (const float*)d_in[14];
  const float* erg2     = (const float*)d_in[15];
  const float* erbb2    = (const float*)d_in[16];
  const float* codebook = (const float*)d_in[17];
  const float* drw1     = (const float*)d_in[18];
  const float* drb1     = (const float*)d_in[19];
  const float* drg1     = (const float*)d_in[20];
  const float* drbb1    = (const float*)d_in[21];
  const float* drw2     = (const float*)d_in[22];
  const float* drb2     = (const float*)d_in[23];
  const float* drg2     = (const float*)d_in[24];
  const float* drbb2    = (const float*)d_in[25];
  const float* dec_w1   = (const float*)d_in[26];
  const float* dec_b1   = (const float*)d_in[27];
  const float* dec_g1   = (const float*)d_in[28];
  const float* dec_bb1  = (const float*)d_in[29];
  const float* dec_w2   = (const float*)d_in[30];
  const float* dec_b2   = (const float*)d_in[31];
  const float* cvar     = (const float*)d_in[32];
  (void)in_sizes; (void)n_in; (void)out_size; (void)ws_size;

  char* ws = (char*)d_ws;
  float*    A0    = (float*)(ws + OFF_ACT0);
  float*    A1    = (float*)(ws + OFF_ACT1);
  float*    A2    = (float*)(ws + OFF_ACT2);
  float*    A3    = (float*)(ws + OFF_ACT3);
  float*    A4    = (float*)(ws + OFF_ACT4);
  float*    A5    = (float*)(ws + OFF_ACT5);
  _Float16* COL   = (_Float16*)(ws + OFF_COL);
  _Float16* WH    = (_Float16*)(ws + OFF_WH);
  _Float16* CBH   = (_Float16*)(ws + OFF_CBH);
  int*      IDX   = (int*)(ws + OFF_IDX);
  float*    STATS = (float*)(ws + OFF_STATS);
  float*    CNT   = (float*)(ws + OFF_CNT);
  float*    CBN   = (float*)(ws + OFF_CBN);
  float*    ACC   = (float*)(ws + OFF_ACC);
  float*    dec   = (float*)d_out;
  float*    tail  = dec + (long)BN_ * CIN_ * IMG_ * IMG_;
  hipStream_t s = stream;

  // zero counts + cbn + accumulators (contiguous region)
  zero_f32_k<<<8, 256, 0, s>>>(CNT, 512 + 512 + 64);

  // ---- encoder conv1: 2->32, k4 s2 p1, out [M1,32] NHWC ----
  cvt_f16_k<<<nblk(1024, 256), 256, 0, s>>>(enc_w1, WH, 32L * 32, 0);
  im2col_k<<<nblk(M1_ * 32, 256), 256, 0, s>>>(inputs, COL, BN_, CIN_, IMG_, IMG_,
                                               64, 64, 4, 4, 2, 1, 1, 0);
  gemm_wmma_bias<<<dim3((unsigned)(M1_ / 256), 2), 128, 0, s>>>(COL, WH, enc_b1, A0, M1_, 32, 32);
  zero_f32_k<<<1, 256, 0, s>>>(STATS, 64);
  bn_stats_k<<<2048, 256, 0, s>>>(A0, STATS, M1_, 32);
  bn_apply_k<<<nblk(M1_ * 32, 256), 256, 0, s>>>(A0, A0, nullptr, STATS, enc_g1, enc_bb1, M1_, 32, 1);

  // ---- encoder conv2: 32->64, k4 s2 p1, out [M2,64] ----
  cvt_f16_k<<<nblk(64L * 512, 256), 256, 0, s>>>(enc_w2, WH, 64L * 512, 0);
  im2col_k<<<nblk(M2_ * 512, 256), 256, 0, s>>>(A0, COL, BN_, 32, 64, 64,
                                                32, 32, 4, 4, 2, 1, 0, 0);
  gemm_wmma_bias<<<dim3((unsigned)(M2_ / 256), 4), 128, 0, s>>>(COL, WH, enc_b2, A1, M2_, 64, 512);
  zero_f32_k<<<1, 256, 0, s>>>(STATS, 128);
  bn_stats_k<<<2048, 256, 0, s>>>(A1, STATS, M2_, 64);
  bn_apply_k<<<nblk(M2_ * 64, 256), 256, 0, s>>>(A1, A1, nullptr, STATS, enc_g2, enc_bb2, M2_, 64, 0);

  // ---- encoder residual blocks ----
  for (int i = 0; i < 2; ++i) {
    run_resblock(s, A1, A2, A3, COL, WH, STATS,
                 erw1 + (size_t)i * RC_ * HC_ * 9, erb1 + i * RC_, erg1 + i * RC_, erbb1 + i * RC_,
                 erw2 + (size_t)i * HC_ * RC_,     erb2 + i * HC_, erg2 + i * HC_, erbb2 + i * HC_);
  }

  // ---- vector quantizer ----
  cb_norm_k<<<8, 64, 0, s>>>(codebook, CBN, KC_, HC_);
  cvt_f16_k<<<nblk((long)KC_ * HC_, 256), 256, 0, s>>>(codebook, CBH, (long)KC_ * HC_, 0);
  cvt_f16_k<<<nblk(M2_ * HC_, 256), 256, 0, s>>>(A1, COL, M2_ * HC_, 0);  // ZH in COL
  vq_argmin_k<<<(unsigned)(M2_ / 64), 128, 0, s>>>(COL, CBH, CBN, IDX);
  vq_gather_k<<<nblk(M2_ * HC_, 256), 256, 0, s>>>(A1, codebook, IDX, A4, CNT, &ACC[0], M2_, HC_);

  // ---- decoder residual blocks (zq == quant numerically) ----
  for (int i = 0; i < 2; ++i) {
    run_resblock(s, A4, A2, A3, COL, WH, STATS,
                 drw1 + (size_t)i * RC_ * HC_ * 9, drb1 + i * RC_, drg1 + i * RC_, drbb1 + i * RC_,
                 drw2 + (size_t)i * HC_ * RC_,     drb2 + i * HC_, drg2 + i * HC_, drbb2 + i * HC_);
  }

  // ---- deconv1: 64->32, out [M1,32] NHWC, then BN + ReLU ----
  deconv_k<<<nblk(M1_ * 32, 256), 256, 0, s>>>(A4, dec_w1, dec_b1, A5, BN_, 32, 32, 64, 32, 0);
  zero_f32_k<<<1, 256, 0, s>>>(STATS, 64);
  bn_stats_k<<<2048, 256, 0, s>>>(A5, STATS, M1_, 32);
  bn_apply_k<<<nblk(M1_ * 32, 256), 256, 0, s>>>(A5, A5, nullptr, STATS, dec_g1, dec_bb1, M1_, 32, 1);

  // ---- deconv2: 32->2, write decoded NCHW directly into d_out ----
  deconv_k<<<nblk((long)BN_ * CIN_ * IMG_ * IMG_, 256), 256, 0, s>>>(
      A5, dec_w2, dec_b2, dec, BN_, 64, 64, 32, CIN_, 1);

  // ---- losses ----
  recon_loss_k<<<nblk((long)BN_ * CIN_ * IMG_ * IMG_, 256), 256, 0, s>>>(
      dec, inputs, cvar, &ACC[1], (long)BN_ * CIN_ * IMG_ * IMG_, CIN_, (long)IMG_ * IMG_);
  finalize_k<<<1, 512, 0, s>>>(CNT, ACC, tail);
}
